// CaseConditionedRefiner_77841987272754
// MI455X (gfx1250) — matrix operand order repaired and checked by reading, not compiled, not run on hardware
//
#include <hip/hip_runtime.h>
#include <math.h>

#define HID    128
#define K1DIM  512
#define NNZD   500000
#define NCASE  50000
#define ETILES (NNZD / 16)   // 31250
#define CTILES (NCASE / 16)  // 3125

typedef __attribute__((ext_vector_type(16))) __bf16 v16bf;
typedef __attribute__((ext_vector_type(8)))  float  v8f;

union BF16x16 {
  v16bf v;
  uint4 u[2];
};

// Hardware f32->bf16 (RNE) via __bf16 cast; lowers to v_cvt_*bf16_f32.
__device__ __forceinline__ unsigned short f2bf(float f) {
  __bf16 h = (__bf16)f;
  return __builtin_bit_cast(unsigned short, h);
}

// ---------------- K0: zero segment accumulators ----------------
__global__ void zero_kernel(float* __restrict__ p, int n) {
  for (int i = blockIdx.x * blockDim.x + threadIdx.x; i < n;
       i += gridDim.x * blockDim.x)
    p[i] = 0.0f;
}

// ---------------- KW: weights -> bf16, transposed (N-major) ----------------
// w1t[n*512+k] = bf16(w1[k][n]); w2t[n*128+k] = bf16(w2[k][n]); cwt likewise.
__global__ void prep_kernel(const float* __restrict__ w1,
                            const float* __restrict__ w2,
                            const float* __restrict__ cw,
                            unsigned short* __restrict__ w1t,
                            unsigned short* __restrict__ w2t,
                            unsigned short* __restrict__ cwt) {
  int i = blockIdx.x * blockDim.x + threadIdx.x;
  if (i < K1DIM * HID) {
    int n = i / K1DIM, k = i % K1DIM;
    w1t[i] = f2bf(w1[k * HID + n]);
  } else if (i < K1DIM * HID + HID * HID) {
    int j = i - K1DIM * HID;
    int n = j / HID, k = j % HID;
    w2t[j] = f2bf(w2[k * HID + n]);
  } else if (i < K1DIM * HID + 2 * HID * HID) {
    int j = i - K1DIM * HID - HID * HID;
    int n = j / HID, k = j % HID;
    cwt[j] = f2bf(cw[k * HID + n]);
  }
}

// ---------------- K1: segment sums via fp32 global atomics ----------------
__global__ __launch_bounds__(256) void scatter_kernel(
    const float* __restrict__ node_repr, const float* __restrict__ edge_vals,
    const int* __restrict__ hpo_idx, const int* __restrict__ case_idx,
    float* __restrict__ accum, float* __restrict__ wsum) {
  int gid = blockIdx.x * 256 + threadIdx.x;
  int e = gid >> 5;
  if (e >= NNZD) return;
  int lane = gid & 31;
  int h = hpo_idx[e];
  int c = case_idx[e];
  float w = edge_vals[e];
  float4 z = *(const float4*)(node_repr + (size_t)h * HID + lane * 4);
  float* dst = accum + (size_t)c * HID + lane * 4;
  unsafeAtomicAdd(dst + 0, w * z.x);
  unsafeAtomicAdd(dst + 1, w * z.y);
  unsafeAtomicAdd(dst + 2, w * z.z);
  unsafeAtomicAdd(dst + 3, w * z.w);
  if (lane == 0) unsafeAtomicAdd(wsum + c, w);
}

// ---------------- K2: case_ctx = sum/max(wsum,eps); ctx_up = case_ctx@ctx_w + b
__global__ __launch_bounds__(256) void ctx_kernel(
    float* __restrict__ accum,            // in: sums, out: case_ctx (f32)
    const float* __restrict__ wsum, const unsigned short* __restrict__ cwt,
    const float* __restrict__ ctx_b, float* __restrict__ ctxup) {
  __shared__ unsigned short actx[16 * HID];  // 4 KB, bf16 A-tile (row-major)
  __shared__ float cup[16 * HID];            // 8 KB, f32 D-tile for coalesced store

  const int tid = threadIdx.x;
  const int lane = tid & 31;
  const int wave = tid >> 5;
  const int n0 = wave * 16;
  const int bn = n0 + (lane & 15);
  const int kband = (lane >> 4) * 16;
  const int m_a = lane & 15;
  const int h_a = lane >> 4;

  BF16x16 cwfr[4];  // B fragments of ctx_w, register resident
#pragma unroll
  for (int kk = 0; kk < 4; ++kk) {
    const uint4* p = (const uint4*)(cwt + (size_t)bn * HID + kk * 32 + kband);
    cwfr[kk].u[0] = p[0];
    cwfr[kk].u[1] = p[1];
  }
  const float biasc = ctx_b[bn];

  const int sm = tid >> 4;
  const int sc = (tid & 15) * 8;

  for (int t = blockIdx.x; t < CTILES; t += gridDim.x) {
    const int case0 = t * 16;
    {
      float* row = accum + (size_t)(case0 + sm) * HID + sc;
      const float inv = 1.0f / fmaxf(wsum[case0 + sm], 1e-8f);
#pragma unroll
      for (int j = 0; j < 8; ++j) {
        float v = row[j] * inv;
        row[j] = v;                      // becomes case_ctx (f32, for gathers)
        actx[sm * HID + sc + j] = f2bf(v);
      }
    }
    __syncthreads();

    v8f acc;
#pragma unroll
    for (int r = 0; r < 8; ++r) acc[r] = biasc;
#pragma unroll
    for (int kk = 0; kk < 4; ++kk) {
      BF16x16 a;
      const uint4* ap = (const uint4*)(actx + m_a * HID + kk * 32 + 16 * h_a);
      a.u[0] = ap[0];   // K = k0 + 16h .. +7
      a.u[1] = ap[1];   // K = k0 + 16h + 8 .. +15
      acc = __builtin_amdgcn_wmma_f32_16x16x32_bf16(
          false, a.v, false, cwfr[kk].v, (short)0, acc, false, false);
    }
#pragma unroll
    for (int r = 0; r < 8; ++r)
      cup[(r + 8 * h_a) * HID + n0 + m_a] = acc[r];
    __syncthreads();

    // coalesced b128 stores of ctx_up
    {
      float* op = ctxup + (size_t)(case0 + sm) * HID + sc;
      float4* ov = (float4*)op;
      const float* src = cup + sm * HID + sc;
      ov[0] = make_float4(src[0], src[1], src[2], src[3]);
      ov[1] = make_float4(src[4], src[5], src[6], src[7]);
    }
    __syncthreads();  // actx/cup WAR vs next tile
  }
}

// ---------------- K3: per-edge MLP gate + mix + LayerNorm ----------------
__global__ __launch_bounds__(256) void refine_kernel(
    const float* __restrict__ node_repr, const int* __restrict__ hpo_idx,
    const int* __restrict__ case_idx, const float* __restrict__ case_ctx,
    const float* __restrict__ ctxup, const unsigned short* __restrict__ w1t,
    const unsigned short* __restrict__ w2t, const float* __restrict__ b1,
    const float* __restrict__ b2, const float* __restrict__ ln_g,
    const float* __restrict__ ln_b, float* __restrict__ out) {
  __shared__ unsigned short gin[16 * K1DIM];  // 16 KB gate_in bf16
  __shared__ unsigned short hbuf[16 * HID];   //  4 KB hidden bf16
  __shared__ float zbuf[16 * HID];            //  8 KB
  __shared__ float cubuf[16 * HID];           //  8 KB
  __shared__ float prebuf[16 * HID];          //  8 KB   (total 44 KB)

  const int tid = threadIdx.x;
  const int lane = tid & 31;
  const int wave = tid >> 5;
  const int n0 = wave * 16;          // this wave's output-column tile
  const int bn = n0 + (lane & 15);   // B-fragment column
  const int kband = (lane >> 4) * 16;
  const int m_a = lane & 15;
  const int h_a = lane >> 4;

  // Register-resident bf16 weight fragments (16*8 + 4*8 VGPRs per wave)
  BF16x16 w1fr[16];
#pragma unroll
  for (int kk = 0; kk < 16; ++kk) {
    const uint4* p = (const uint4*)(w1t + (size_t)bn * K1DIM + kk * 32 + kband);
    w1fr[kk].u[0] = p[0];
    w1fr[kk].u[1] = p[1];
  }
  BF16x16 w2fr[4];
#pragma unroll
  for (int kk = 0; kk < 4; ++kk) {
    const uint4* p = (const uint4*)(w2t + (size_t)bn * HID + kk * 32 + kband);
    w2fr[kk].u[0] = p[0];
    w2fr[kk].u[1] = p[1];
  }
  const float bias1 = b1[bn];
  const float bias2 = b2[bn];

  const int sm = tid >> 4;         // staging row (0..15)
  const int sc = (tid & 15) * 8;   // staging col start

  for (int t = blockIdx.x; t < ETILES; t += gridDim.x) {
    const int e0 = t * 16;

    // ---- stage z / ctx / ctx_up gathers, build gate_in bf16 ----
    {
      const int e = e0 + sm;
      const int hr = hpo_idx[e];
      const int cr = case_idx[e];
      const float4* zp = (const float4*)(node_repr + (size_t)hr * HID + sc);
      const float4* cp = (const float4*)(case_ctx + (size_t)cr * HID + sc);
      const float4* up = (const float4*)(ctxup + (size_t)cr * HID + sc);
      float z[8], cx[8], cu[8];
      float4 a0 = zp[0], a1 = zp[1];
      z[0]=a0.x; z[1]=a0.y; z[2]=a0.z; z[3]=a0.w;
      z[4]=a1.x; z[5]=a1.y; z[6]=a1.z; z[7]=a1.w;
      float4 b0 = cp[0], b1q = cp[1];
      cx[0]=b0.x; cx[1]=b0.y; cx[2]=b0.z; cx[3]=b0.w;
      cx[4]=b1q.x; cx[5]=b1q.y; cx[6]=b1q.z; cx[7]=b1q.w;
      float4 c0q = up[0], c1q = up[1];
      cu[0]=c0q.x; cu[1]=c0q.y; cu[2]=c0q.z; cu[3]=c0q.w;
      cu[4]=c1q.x; cu[5]=c1q.y; cu[6]=c1q.z; cu[7]=c1q.w;
#pragma unroll
      for (int j = 0; j < 8; ++j) {
        zbuf[sm * HID + sc + j] = z[j];
        cubuf[sm * HID + sc + j] = cu[j];
        gin[sm * K1DIM + sc + j]           = f2bf(z[j]);
        gin[sm * K1DIM + HID + sc + j]     = f2bf(cx[j]);
        gin[sm * K1DIM + 2 * HID + sc + j] = f2bf(z[j] * cx[j]);
        gin[sm * K1DIM + 3 * HID + sc + j] = f2bf(fabsf(z[j] - cx[j]));
      }
    }
    __syncthreads();

    // ---- GEMM1: h = relu(gate_in @ w1 + b1), K = 512 (16 WMMA steps) ----
    v8f acc;
#pragma unroll
    for (int r = 0; r < 8; ++r) acc[r] = bias1;
#pragma unroll
    for (int kk = 0; kk < 16; ++kk) {
      BF16x16 a;
      const uint4* ap = (const uint4*)(gin + m_a * K1DIM + kk * 32 + 16 * h_a);
      a.u[0] = ap[0];
      a.u[1] = ap[1];
      acc = __builtin_amdgcn_wmma_f32_16x16x32_bf16(
          false, a.v, false, w1fr[kk].v, (short)0, acc, false, false);
    }
#pragma unroll
    for (int r = 0; r < 8; ++r)
      hbuf[(r + 8 * h_a) * HID + n0 + m_a] = f2bf(fmaxf(acc[r], 0.0f));
    __syncthreads();

    // ---- GEMM2: gate = sigmoid(h @ w2 + b2); gate mix + residual ----
    v8f acc2;
#pragma unroll
    for (int r = 0; r < 8; ++r) acc2[r] = bias2;
#pragma unroll
    for (int kk = 0; kk < 4; ++kk) {
      BF16x16 a;
      const uint4* ap = (const uint4*)(hbuf + m_a * HID + kk * 32 + 16 * h_a);
      a.u[0] = ap[0];
      a.u[1] = ap[1];
      acc2 = __builtin_amdgcn_wmma_f32_16x16x32_bf16(
          false, a.v, false, w2fr[kk].v, (short)0, acc2, false, false);
    }
#pragma unroll
    for (int r = 0; r < 8; ++r) {
      const int mr = r + 8 * h_a;
      const int c = n0 + m_a;
      const float g = 1.0f / (1.0f + __expf(-acc2[r]));
      const float zv = zbuf[mr * HID + c];
      const float cu = cubuf[mr * HID + c];
      const float upd = (1.0f - g) * zv + g * cu;
      prebuf[mr * HID + c] = 0.7f * zv + 0.3f * upd;
    }
    __syncthreads();

    // ---- LayerNorm over 128 channels (16 lanes/row, shfl_xor reduce) ----
    {
      float v[8];
      float s = 0.0f, ss = 0.0f;
#pragma unroll
      for (int j = 0; j < 8; ++j) {
        v[j] = prebuf[sm * HID + sc + j];
        s += v[j];
        ss += v[j] * v[j];
      }
#pragma unroll
      for (int o = 1; o < 16; o <<= 1) {
        s += __shfl_xor(s, o, 32);
        ss += __shfl_xor(ss, o, 32);
      }
      const float mu = s * (1.0f / HID);
      const float var = ss * (1.0f / HID) - mu * mu;
      const float inv = rsqrtf(var + 1e-5f);
      float r0[8];
#pragma unroll
      for (int j = 0; j < 8; ++j)
        r0[j] = (v[j] - mu) * inv * ln_g[sc + j] + ln_b[sc + j];
      float4* ov = (float4*)(out + (size_t)(e0 + sm) * HID + sc);
      ov[0] = make_float4(r0[0], r0[1], r0[2], r0[3]);
      ov[1] = make_float4(r0[4], r0[5], r0[6], r0[7]);
    }
    // no trailing barrier needed: next tile's first write to prebuf is two
    // barriers downstream of every thread's LN reads
  }
}

// ---------------- host-side launcher ----------------
extern "C" void kernel_launch(void* const* d_in, const int* in_sizes, int n_in,
                              void* d_out, int out_size, void* d_ws,
                              size_t ws_size, hipStream_t stream) {
  const float* node_repr = (const float*)d_in[0];
  const float* ctx_w     = (const float*)d_in[1];
  const float* ctx_b     = (const float*)d_in[2];
  const float* w1        = (const float*)d_in[3];
  const float* b1        = (const float*)d_in[4];
  const float* w2        = (const float*)d_in[5];
  const float* b2        = (const float*)d_in[6];
  const float* ln_g      = (const float*)d_in[7];
  const float* ln_b      = (const float*)d_in[8];
  const float* edge_vals = (const float*)d_in[9];
  const int*   hpo_idx   = (const int*)d_in[10];
  const int*   case_idx  = (const int*)d_in[11];
  float* out = (float*)d_out;

  // workspace layout (~51.6 MB total)
  char* ws = (char*)d_ws;
  float* accum = (float*)(ws + 0);                         // 50000*128 f32 -> case_ctx
  float* wsum  = (float*)(ws + 25600000);                  // 50000 f32 (contiguous)
  float* ctxup = (float*)(ws + 25800192);                  // 50000*128 f32
  unsigned short* w1t = (unsigned short*)(ws + 51400192);  // 512*128 bf16
  unsigned short* w2t = (unsigned short*)(ws + 51531264);  // 128*128 bf16
  unsigned short* cwt = (unsigned short*)(ws + 51564032);  // 128*128 bf16

  zero_kernel<<<2048, 256, 0, stream>>>(accum, NCASE * HID + NCASE);
  prep_kernel<<<(K1DIM * HID + 2 * HID * HID + 255) / 256, 256, 0, stream>>>(
      w1, w2, ctx_w, w1t, w2t, cwt);
  scatter_kernel<<<(NNZD * 32) / 256, 256, 0, stream>>>(
      node_repr, edge_vals, hpo_idx, case_idx, accum, wsum);
  ctx_kernel<<<1024, 256, 0, stream>>>(accum, wsum, cwt, ctx_b, ctxup);
  refine_kernel<<<2048, 256, 0, stream>>>(node_repr, hpo_idx, case_idx, accum,
                                          ctxup, w1t, w2t, b1, b2, ln_g, ln_b,
                                          out);
}
